// LocalFeatureFusion_592705487298
// MI455X (gfx1250) — matrix operand order repaired and checked by reading, not compiled, not run on hardware
//
#include <hip/hip_runtime.h>
#include <hip/hip_bf16.h>

typedef __attribute__((ext_vector_type(16))) _Float16 v16h;
typedef __attribute__((ext_vector_type(8)))  _Float16 v8h;
typedef __attribute__((ext_vector_type(8)))  float    v8f;

#define BN    16384   // B*N groups
#define NQ    4096
#define MK    4096
#define CC    256
#define KNN   8
#define SPAD  16      // tokens per group padded 9 -> 16
#define FFD   1024
#define RAD2  0.0036f // 0.06^2
#define GPB   4       // groups per block
#define ROWS  (GPB*SPAD)  // 64
#define SX    264     // LDS halfword stride for 256-col buffers
#define SSTR  1032    // LDS halfword stride for 1024-col buffer
#define SG    260     // LDS float stride for fp32 gather staging

// ---------------------------------------------------------------------------
// Pack weight W[N,K] (torch layout, y = x @ W.T) into WMMA f16 B-fragment
// order: tile (kt,nt) -> 32 lanes x 16 contiguous halves.
// lane<16 : n = nt*16+lane,    k = kt*32 + j      (j=0..15)
// lane>=16: n = nt*16+lane-16, k = kt*32 + 16 + j
// ---------------------------------------------------------------------------
__global__ void pack_w(const float* __restrict__ W, _Float16* __restrict__ out,
                       int Kd, int Nd) {
  int e = blockIdx.x * 256 + threadIdx.x;
  if (e >= Kd * Nd) return;
  int j    = e & 15;
  int lane = (e >> 4) & 31;
  int tile = e >> 9;
  int Ntiles = Nd >> 4;
  int nt = tile % Ntiles, kt = tile / Ntiles;
  int n = nt * 16 + (lane & 15);
  int k = kt * 32 + ((lane >> 4) << 4) + j;
  out[e] = (_Float16)W[(size_t)n * Kd + k];
}

// ---------------------------------------------------------------------------
// KNN: one wave32 per query. Per-lane sorted top-8, merged across lanes with
// shfl_xor min-reductions (tie-break on smaller index, matching top_k).
// ---------------------------------------------------------------------------
__global__ __launch_bounds__(256) void knn_kernel(
    const float* __restrict__ qxyz, const float* __restrict__ kxyz,
    const unsigned char* __restrict__ kpad, int* __restrict__ idx_out) {
  int wave = threadIdx.x >> 5, lane = threadIdx.x & 31;
  int qi = blockIdx.x * 8 + wave;
  int b  = qi >> 12;           // qi / 4096
  const float* qp = qxyz + (size_t)qi * 3;
  float qx = qp[0], qy = qp[1], qz = qp[2];

  float bd[8]; int bi[8];
#pragma unroll
  for (int i = 0; i < 8; ++i) { bd[i] = 3.4e38f; bi[i] = -1; }

  const float* kb = kxyz + (size_t)b * MK * 3;
  const unsigned char* pb = kpad + (size_t)b * MK;
  for (int m = lane; m < MK; m += 32) {
    float dx = qx - kb[m*3+0], dy = qy - kb[m*3+1], dz = qz - kb[m*3+2];
    float d2 = dx*dx + dy*dy + dz*dz;
    if (pb[m] || d2 > RAD2) continue;
    if (d2 < bd[7]) {
      bd[7] = d2; bi[7] = m;
#pragma unroll
      for (int j = 7; j > 0; --j) {
        if (bd[j] < bd[j-1]) {
          float td = bd[j]; bd[j] = bd[j-1]; bd[j-1] = td;
          int   ti = bi[j]; bi[j] = bi[j-1]; bi[j-1] = ti;
        }
      }
    }
  }
  int head = 0;
  for (int k = 0; k < 8; ++k) {
    float wd = (head < 8) ? bd[head] : 3.4e38f;
    int   wi = (head < 8) ? bi[head] : -1;
    for (int off = 16; off; off >>= 1) {
      float od = __shfl_xor(wd, off, 32);
      int   oi = __shfl_xor(wi, off, 32);
      if (od < wd || (od == wd && (unsigned)oi < (unsigned)wi)) { wd = od; wi = oi; }
    }
    if (wi >= 0 && head < 8 && bi[head] == wi) head++;
    if (lane == 0) idx_out[(size_t)qi * 8 + k] = (wd < 3.0e38f) ? wi : -1;
  }
}

// ---------------------------------------------------------------------------
// WMMA GEMM: one N-tile per wave iteration, 4 M-tiles (64 rows) accumulated
// simultaneously -> each B fragment (from packed L2 weights) feeds 4 WMMAs.
// ---------------------------------------------------------------------------
__device__ __forceinline__ void gemm_lds(
    const _Float16* A, int sa, const _Float16* __restrict__ Bp,
    const float* __restrict__ bias, _Float16* O, int so,
    int Kd, int Ntiles, int relu) {
  int wave = threadIdx.x >> 5, lane = threadIdx.x & 31;
  int col = lane & 15, half = lane >> 4;
  int ktiles = Kd >> 5;
  for (int nt = wave; nt < Ntiles; nt += 8) {
    float bv = bias ? bias[nt * 16 + col] : 0.0f;
    v8f acc[4];
#pragma unroll
    for (int mt = 0; mt < 4; ++mt)
#pragma unroll
      for (int i = 0; i < 8; ++i) acc[mt][i] = bv;

    const _Float16* bptr = Bp + ((size_t)nt * 32 + lane) * 16;
    const size_t bstride = (size_t)Ntiles * 512;   // 32 lanes * 16 halves
    for (int kt = 0; kt < ktiles; ++kt) {
      v8h b0 = *(const v8h*)(bptr);
      v8h b1 = *(const v8h*)(bptr + 8);
      v16h bf = __builtin_shufflevector(b0, b1, 0,1,2,3,4,5,6,7,8,9,10,11,12,13,14,15);
      bptr += bstride;
      const _Float16* ab = A + (size_t)col * sa + half * 16 + (size_t)kt * 32;
#pragma unroll
      for (int mt = 0; mt < 4; ++mt) {
        v8h a0 = *(const v8h*)(ab + (size_t)mt * 16 * sa);
        v8h a1 = *(const v8h*)(ab + (size_t)mt * 16 * sa + 8);
        v16h af = __builtin_shufflevector(a0, a1, 0,1,2,3,4,5,6,7,8,9,10,11,12,13,14,15);
        acc[mt] = __builtin_amdgcn_wmma_f32_16x16x32_f16(
            false, af, false, bf, (short)0, acc[mt], false, false);
      }
    }
#pragma unroll
    for (int mt = 0; mt < 4; ++mt) {
      _Float16* orow = O + (size_t)(mt * 16 + half * 8) * so + nt * 16 + col;
#pragma unroll
      for (int i = 0; i < 8; ++i) {
        float v = acc[mt][i];
        if (relu) v = fmaxf(v, 0.0f);
        orow[(size_t)i * so] = (_Float16)v;
      }
    }
  }
}

// Residual add + LayerNorm over 256 cols, f16 in / f16 out.
__device__ __forceinline__ void ln_res(
    const _Float16* Xa, int sxa, const _Float16* Xb, int sxb,
    const float* __restrict__ g, const float* __restrict__ be,
    _Float16* O, int so) {
  int wave = threadIdx.x >> 5, lane = threadIdx.x & 31;
  for (int r = wave; r < ROWS; r += 8) {
    const _Float16* pa = Xa + (size_t)r * sxa;
    const _Float16* pb = Xb + (size_t)r * sxb;
    float vals[8], s = 0.0f, sq = 0.0f;
#pragma unroll
    for (int i = 0; i < 8; ++i) {
      int c = lane * 8 + i;
      float v = (float)pa[c] + (float)pb[c];
      vals[i] = v; s += v; sq += v * v;
    }
    for (int off = 16; off; off >>= 1) {
      s  += __shfl_xor(s,  off, 32);
      sq += __shfl_xor(sq, off, 32);
    }
    float mu   = s * (1.0f / 256.0f);
    float var  = sq * (1.0f / 256.0f) - mu * mu;
    float rinv = rsqrtf(var + 1e-5f);
    _Float16* po = O + (size_t)r * so;
#pragma unroll
    for (int i = 0; i < 8; ++i) {
      int c = lane * 8 + i;
      po[c] = (_Float16)((vals[i] - mu) * rinv * g[c] + be[c]);
    }
  }
}

// ---------------------------------------------------------------------------
// Fused transformer block: 1 workgroup = 4 groups = 64 padded token rows.
// ---------------------------------------------------------------------------
__global__ __launch_bounds__(256) void fuse_kernel(
    const float* __restrict__ qfeat, const float* __restrict__ kvfeat,
    const int* __restrict__ idx,
    const _Float16* __restrict__ Wqkv_p, const float* __restrict__ bqkv,
    const _Float16* __restrict__ Wo_p,   const float* __restrict__ bo,
    const _Float16* __restrict__ W1_p,   const float* __restrict__ b1,
    const _Float16* __restrict__ W2_p,   const float* __restrict__ b2,
    const float* __restrict__ g1, const float* __restrict__ be1,
    const float* __restrict__ g2, const float* __restrict__ be2,
    float* __restrict__ out) {
  __shared__ _Float16 ldsX[ROWS * SX];    // tokens (f16), later FFN output f
  __shared__ _Float16 ldsY[ROWS * SX];    // o-proj out, later x = LN1
  __shared__ _Float16 ldsS[ROWS * SSTR];  // QKV(768) + attn-o(768..1023); later FFN hidden
  __shared__ float    stage[ROWS * SG];   // fp32 async-gather staging
  __shared__ float    padb[ROWS];         // 0 valid, -1e30 masked (key bias)

  int tid  = threadIdx.x;
  int wave = tid >> 5, lane = tid & 31;
  int g0   = blockIdx.x * GPB;

  // L2 warm hints for the packed weights
  __builtin_prefetch(Wqkv_p + (size_t)tid * 64, 0, 1);
  __builtin_prefetch(W1_p   + (size_t)tid * 64, 0, 1);

  // ---- stage A: async gather of token rows (fp32) into LDS staging ----
  // wave w owns rows w, w+8, ... ; each lane async-copies 32B of the 1KB row.
  for (int r = wave; r < ROWS; r += 8) {
    int g = r >> 4, s = r & 15;
    int gi = g0 + g;
    int b  = gi >> 12;
    const float* src = nullptr;
    float pbias = 0.0f;
    if (s == 0) {
      src = qfeat + (size_t)gi * CC;
    } else if (s <= KNN) {
      int j = idx[(size_t)gi * KNN + (s - 1)];
      if (j < 0) { src = qfeat + (size_t)gi * CC; pbias = -1e30f; }
      else        src = kvfeat + ((size_t)b * MK + j) * CC;
    } else {
      pbias = -1e30f;   // padded token slot
    }
    if (lane == 0) padb[r] = pbias;
    if (src) {
      unsigned int ldsoff =
          (unsigned int)(unsigned long long)(const void*)&stage[(size_t)r * SG]
          + (unsigned int)(lane * 32);
      unsigned long long gaddr = (unsigned long long)(src + lane * 8);
      asm volatile(
          "global_load_async_to_lds_b128 %0, %1, off\n\t"
          "global_load_async_to_lds_b128 %0, %1, off offset:16"
          :: "v"(ldsoff), "v"(gaddr) : "memory");
    }
  }
  asm volatile("s_wait_asynccnt 0x0" ::: "memory");
  __syncthreads();

  // ---- convert staged fp32 rows -> f16 token matrix ldsX ----
  for (int r = wave; r < ROWS; r += 8) {
    int s = r & 15;
    const float* st = &stage[(size_t)r * SG];
#pragma unroll
    for (int i = 0; i < 8; ++i) {
      int c = lane + i * 32;
      ldsX[(size_t)r * SX + c] = (s > KNN) ? (_Float16)0.0f : (_Float16)st[c];
    }
  }
  __syncthreads();

  // ---- QKV projection: X[64x256] @ Wqkv.T -> ldsS cols [0,768) ----
  gemm_lds(ldsX, SX, Wqkv_p, bqkv, ldsS, SSTR, CC, 48, 0);
  __syncthreads();

  // ---- attention (per group, per head) in VALU; o -> ldsS cols [768,1024) ----
  for (int p = wave; p < GPB * 8; p += 8) {
    int g = p >> 3, h = p & 7;
    if (lane < 16) {
      int row = g * 16 + lane;
      const _Float16* Qr = ldsS + (size_t)row * SSTR + h * 32;
      float qv[32];
#pragma unroll
      for (int d = 0; d < 32; ++d) qv[d] = (float)Qr[d];
      float sc[16], mx = -3.4e38f;
#pragma unroll
      for (int t = 0; t < 16; ++t) {
        const _Float16* Kr = ldsS + (size_t)(g*16 + t) * SSTR + 256 + h * 32;
        float a = 0.0f;
#pragma unroll
        for (int d = 0; d < 32; ++d) a += qv[d] * (float)Kr[d];
        a = a * 0.17677669529f + padb[g*16 + t];   // 1/sqrt(32)
        sc[t] = a; mx = fmaxf(mx, a);
      }
      float den = 0.0f;
#pragma unroll
      for (int t = 0; t < 16; ++t) { sc[t] = __expf(sc[t] - mx); den += sc[t]; }
      float rden = 1.0f / den;
      _Float16* Or = ldsS + (size_t)row * SSTR + 768 + h * 32;
#pragma unroll
      for (int d = 0; d < 32; ++d) {
        float o = 0.0f;
#pragma unroll
        for (int t = 0; t < 16; ++t)
          o += sc[t] * (float)ldsS[(size_t)(g*16 + t) * SSTR + 512 + h * 32 + d];
        Or[d] = (_Float16)(o * rden);
      }
    }
  }
  __syncthreads();

  // ---- output projection: o @ Wo.T -> ldsY ----
  gemm_lds(ldsS + 768, SSTR, Wo_p, bo, ldsY, SX, CC, 16, 0);
  __syncthreads();

  // ---- x = LN1(tokens + o_proj) -> ldsY ----
  ln_res(ldsX, SX, ldsY, SX, g1, be1, ldsY, SX);
  __syncthreads();

  // ---- FFN1: relu(x @ W1.T + b1) -> ldsS [64x1024] ----
  gemm_lds(ldsY, SX, W1_p, b1, ldsS, SSTR, CC, 64, 1);
  __syncthreads();

  // ---- FFN2: hidden @ W2.T + b2 -> ldsX (tokens dead) ----
  gemm_lds(ldsS, SSTR, W2_p, b2, ldsX, SX, FFD, 16, 0);
  __syncthreads();

  // ---- x2 = LN2(x + f) -> ldsX ----
  ln_res(ldsY, SX, ldsX, SX, g2, be2, ldsX, SX);
  __syncthreads();

  // ---- emit token 0 of each group + q_feat residual ----
  for (int e = tid; e < GPB * CC; e += 256) {
    int g = e >> 8, c = e & 255;
    int gi = g0 + g;
    out[(size_t)gi * CC + c] =
        (float)ldsX[(size_t)(g * 16) * SX + c] + qfeat[(size_t)gi * CC + c];
  }
}

// ---------------------------------------------------------------------------
extern "C" void kernel_launch(void* const* d_in, const int* in_sizes, int n_in,
                              void* d_out, int out_size, void* d_ws, size_t ws_size,
                              hipStream_t stream) {
  (void)in_sizes; (void)n_in; (void)out_size; (void)ws_size;
  const float* q_xyz  = (const float*)d_in[0];
  const float* q_feat = (const float*)d_in[1];
  const float* kv_xyz = (const float*)d_in[2];
  const float* kv_feat= (const float*)d_in[3];
  const unsigned char* kv_pad = (const unsigned char*)d_in[4];
  const float* Wqkv = (const float*)d_in[5];
  const float* bqkv = (const float*)d_in[6];
  const float* Wo   = (const float*)d_in[7];
  const float* bo   = (const float*)d_in[8];
  const float* W1   = (const float*)d_in[9];
  const float* b1   = (const float*)d_in[10];
  const float* W2   = (const float*)d_in[11];
  const float* b2   = (const float*)d_in[12];
  const float* g1   = (const float*)d_in[13];
  const float* be1  = (const float*)d_in[14];
  const float* g2   = (const float*)d_in[15];
  const float* be2  = (const float*)d_in[16];

  char* ws = (char*)d_ws;
  int* idx = (int*)ws;
  size_t off = (size_t)BN * KNN * sizeof(int);          // 512 KB
  _Float16* Wqkv_p = (_Float16*)(ws + off); off += (size_t)768  * 256  * 2;
  _Float16* Wo_p   = (_Float16*)(ws + off); off += (size_t)256  * 256  * 2;
  _Float16* W1_p   = (_Float16*)(ws + off); off += (size_t)1024 * 256  * 2;
  _Float16* W2_p   = (_Float16*)(ws + off); off += (size_t)1024 * 256  * 2;

  pack_w<<<(768  * 256 + 255) / 256, 256, 0, stream>>>(Wqkv, Wqkv_p, 256,  768);
  pack_w<<<(256  * 256 + 255) / 256, 256, 0, stream>>>(Wo,   Wo_p,   256,  256);
  pack_w<<<(1024 * 256 + 255) / 256, 256, 0, stream>>>(W1,   W1_p,   256,  1024);
  pack_w<<<(1024 * 256 + 255) / 256, 256, 0, stream>>>(W2,   W2_p,   1024, 256);

  knn_kernel<<<BN / 8, 256, 0, stream>>>(q_xyz, kv_xyz, kv_pad, idx);

  fuse_kernel<<<BN / GPB, 256, 0, stream>>>(
      q_feat, kv_feat, idx,
      Wqkv_p, bqkv, Wo_p, bo, W1_p, b1, W2_p, b2,
      g1, be1, g2, be2, (float*)d_out);
}
